// MultimodalGNN_76089640616127
// MI455X (gfx1250) — matrix operand order repaired and checked by reading, not compiled, not run on hardware
//
#include <hip/hip_runtime.h>
#include <hip/hip_bf16.h>

typedef __attribute__((ext_vector_type(16))) _Float16 v16h;
typedef __attribute__((ext_vector_type(8)))  _Float16 v8h;
typedef __attribute__((ext_vector_type(8)))  float    v8f;

#define N_IMG   16          // BS*F
#define C0      64
#define NNODES  36
#define NENT    134
#define NEDGE   14

// ---------------------------------------------------------------------------
// NCHW f32 (a+b) -> NHWC f16 via LDS transpose. C fixed at 64.
// Block: 256 threads, one img (blockIdx.y), 64 pixels (blockIdx.x).
// ---------------------------------------------------------------------------
__global__ void k_addcast_nhwc(const float* __restrict__ a, const float* __restrict__ b,
                               _Float16* __restrict__ out, int HW) {
    __shared__ _Float16 tile[64][65];
    const int img = blockIdx.y;
    const int p0 = blockIdx.x * 64;
    const int t = threadIdx.x;
    const int cq = t >> 6, pl = t & 63;
    #pragma unroll
    for (int i = 0; i < 16; ++i) {
        const int c = cq * 16 + i;
        const size_t idx = ((size_t)img * 64 + c) * HW + p0 + pl;
        tile[c][pl] = (_Float16)(a[idx] + b[idx]);
    }
    __syncthreads();
    const int c2 = t & 63, pq = t >> 6;
    #pragma unroll
    for (int i = 0; i < 16; ++i) {
        const int p = pq * 16 + i;
        out[((size_t)img * HW + p0 + p) * 64 + c2] = tile[c2][p];
    }
}

// ---------------------------------------------------------------------------
// Weight prepack into WMMA A-fragment layout (wave32, 16x16x32 f16).
// Apack index: (((mtile*9 + tap)*cchunks + cc)*32 + lane)*16 halves.
// Per lane (half = lane>>4, m16 = lane&15): element e maps to
//   K = 16*half + e   (contiguous, matches B loads)
// MODE 0: forward conv, W layout (Cout, Cin, 3, 3)
// MODE 1: transposed conv, W layout (Cin, Cout, 3, 3)
// ---------------------------------------------------------------------------
template<int MODE>
__global__ void k_prepack(const float* __restrict__ W, _Float16* __restrict__ Apack,
                          int Cin, int Cout) {
    const int lane = threadIdx.x;
    const int half = lane >> 4, m16 = lane & 15;
    const int cchunks = Cin >> 5;
    int x = blockIdx.x;
    const int cc = x % cchunks; x /= cchunks;
    const int tap = x % 9; const int mtile = x / 9;
    const int kh = tap / 3, kw = tap % 3;
    const int m = mtile * 16 + m16;
    v16h frag;
    #pragma unroll
    for (int e = 0; e < 16; ++e) {
        const int j = e >> 1, t = e & 1;
        const int kic = 16 * half + 2 * j + t;
        const int cin = cc * 32 + kic;
        float w;
        if (MODE == 0) w = W[(((size_t)m * Cin + cin) * 3 + kh) * 3 + kw];
        else           w = W[(((size_t)cin * Cout + m) * 3 + kh) * 3 + kw];
        frag[e] = (_Float16)w;
    }
    *(v16h*)(Apack + ((size_t)blockIdx.x * 32 + lane) * 16) = frag;
}

// ---------------------------------------------------------------------------
// Implicit-GEMM conv via WMMA, NHWC activations, 2M x 4N register blocking.
// One wave computes a 32(ch) x 64(pixel) output tile; 8 WMMAs per K-chunk
// reuse 2 A-fragments and 4 B-fragments (all single 32B vector loads).
// MODE 0: conv k3 s2 p1.  MODE 1: convT k3 s2 p1 op1 (invalid taps -> zeros).
// Epilogue: y = relu(acc + bias); y += resA (+resB) post-ReLU.
// OUT_NCHW=1: final layer, strided NCHW stores + NCHW residuals (d_out layout).
// ---------------------------------------------------------------------------
template<int MODE, int OUT_NCHW>
__global__ __launch_bounds__(32)
void k_conv_wmma(const _Float16* __restrict__ Bsrc, const _Float16* __restrict__ Apack,
                 const float* __restrict__ bias,
                 const float* __restrict__ resA, const float* __restrict__ resB,
                 float* __restrict__ outF, _Float16* __restrict__ outH,
                 int Cin, int Cout, int Hin, int Win, int Hout, int Wout) {
    const int lane = threadIdx.x;
    const int half = lane >> 4, l16 = lane & 15;
    const int img = blockIdx.z;
    const int n0 = blockIdx.x * 64;
    const int cchunks = Cin >> 5;
    const size_t HWin = (size_t)Hin * Win;
    const _Float16* in_img = Bsrc + (size_t)img * HWin * Cin;

    v8f zero = {};
    v8f acc[2][4];
    #pragma unroll
    for (int m = 0; m < 2; ++m)
        #pragma unroll
        for (int s = 0; s < 4; ++s) acc[m][s] = zero;

    const _Float16* ap0 = Apack + ((size_t)(blockIdx.y * 2    ) * 9 * cchunks) * 512
                                + (size_t)lane * 16;
    const _Float16* ap1 = Apack + ((size_t)(blockIdx.y * 2 + 1) * 9 * cchunks) * 512
                                + (size_t)lane * 16;

    int oys[4], oxs[4];
    #pragma unroll
    for (int s = 0; s < 4; ++s) {
        const int n = n0 + s * 16 + l16;
        oys[s] = n / Wout; oxs[s] = n % Wout;
    }

    for (int tap = 0; tap < 9; ++tap) {
        const int kh = tap / 3, kw = tap % 3;
        const _Float16* bptr[4]; bool bval[4];
        #pragma unroll
        for (int s = 0; s < 4; ++s) {
            int iy, ix; bool valid;
            if (MODE == 0) {
                iy = 2 * oys[s] - 1 + kh; ix = 2 * oxs[s] - 1 + kw;
                valid = (iy >= 0) && (iy < Hin) && (ix >= 0) && (ix < Win);
            } else {
                const int ty = oys[s] + 1 - kh, tx = oxs[s] + 1 - kw;
                valid = ((ty & 1) == 0) && ((tx & 1) == 0);
                iy = ty >> 1; ix = tx >> 1;
                valid = valid && (iy >= 0) && (iy < Hin) && (ix >= 0) && (ix < Win);
            }
            bval[s] = valid;
            const int iyc = valid ? iy : 0, ixc = valid ? ix : 0;
            bptr[s] = in_img + ((size_t)iyc * Win + ixc) * Cin + 16 * half;
        }
        for (int cc = 0; cc < cchunks; ++cc) {
            v16h a0 = *(const v16h*)ap0; ap0 += 512;
            v16h a1 = *(const v16h*)ap1; ap1 += 512;
            #pragma unroll
            for (int s = 0; s < 4; ++s) {
                v16h b = {};
                if (bval[s]) b = *(const v16h*)(bptr[s] + cc * 32);
                acc[0][s] = __builtin_amdgcn_wmma_f32_16x16x32_f16(
                                false, a0, false, b, (short)0, acc[0][s], false, false);
                acc[1][s] = __builtin_amdgcn_wmma_f32_16x16x32_f16(
                                false, a1, false, b, (short)0, acc[1][s], false, false);
            }
        }
    }

    // epilogue: D VGPR r -> (M = r + 8*half, N = lane&15)
    const size_t HWo = (size_t)Hout * Wout;
    #pragma unroll
    for (int s = 0; s < 4; ++s) {
        const int n = n0 + s * 16 + l16;
        #pragma unroll
        for (int mb = 0; mb < 2; ++mb) {
            const int ch0 = (blockIdx.y * 2 + mb) * 16 + 8 * half;
            float y[8];
            #pragma unroll
            for (int r = 0; r < 8; ++r) {
                float v = acc[mb][s][r] + bias[ch0 + r];
                y[r] = v > 0.f ? v : 0.f;
            }
            if (OUT_NCHW) {
                const size_t base = (size_t)img * Cout * HWo + n;
                #pragma unroll
                for (int r = 0; r < 8; ++r) {
                    const size_t idx = base + (size_t)(ch0 + r) * HWo;
                    float v = y[r];
                    if (resA) v += resA[idx];
                    if (resB) v += resB[idx];
                    outF[idx] = v;
                }
            } else {
                const size_t base = ((size_t)img * HWo + n) * Cout + ch0;
                if (resA) {
                    #pragma unroll
                    for (int r = 0; r < 8; ++r) y[r] += resA[base + r];
                }
                if (outF) {
                    float4 lo = make_float4(y[0], y[1], y[2], y[3]);
                    float4 hi = make_float4(y[4], y[5], y[6], y[7]);
                    *(float4*)(outF + base) = lo;
                    *(float4*)(outF + base + 4) = hi;
                }
                if (outH) {
                    v8h h8;
                    #pragma unroll
                    for (int r = 0; r < 8; ++r) h8[r] = (_Float16)y[r];
                    *(v8h*)(outH + base) = h8;
                }
            }
        }
    }
}

// ---------------------------------------------------------------------------
// NHWC spatial max-pool + LayerNorm over 256 channels. One block per image.
// ---------------------------------------------------------------------------
__global__ void k_pool_ln(const float* __restrict__ in, float* __restrict__ out, int HW) {
    __shared__ float red[256];
    const int img = blockIdx.x, c = threadIdx.x;
    const float* p = in + (size_t)img * HW * 256 + c;
    float m = -1e30f;
    for (int i = 0; i < HW; ++i) m = fmaxf(m, p[(size_t)i * 256]);
    red[c] = m; __syncthreads();
    for (int s = 128; s > 0; s >>= 1) { if (c < s) red[c] += red[c + s]; __syncthreads(); }
    const float mu = red[0] * (1.f / 256.f); __syncthreads();
    red[c] = (m - mu) * (m - mu); __syncthreads();
    for (int s = 128; s > 0; s >>= 1) { if (c < s) red[c] += red[c + s]; __syncthreads(); }
    const float var = red[0] * (1.f / 256.f);
    out[(size_t)img * 256 + c] = (m - mu) * rsqrtf(var + 1e-5f);
}

// text pooling: tn[b][ch*4+p] = max_q (tf+tp)[b][ch][p*4+q]
__global__ void k_tn(const float* __restrict__ tf, const float* __restrict__ tp,
                     float* __restrict__ tn) {
    const int idx = blockIdx.x * blockDim.x + threadIdx.x;
    if (idx >= 2 * 256) return;
    const int b = idx >> 8, q = idx & 255, ch = q >> 2, p = q & 3;
    float m = -1e30f;
    for (int t = 0; t < 4; ++t) {
        const size_t i = ((size_t)b * 64 + ch) * 16 + p * 4 + t;
        m = fmaxf(m, tf[i] + tp[i]);
    }
    tn[idx] = m;
}

// node feature assembly: [vb(8), db(8), tn, obs] per batch
__global__ void k_build_x(const float* __restrict__ vn, const float* __restrict__ dn,
                          const float* __restrict__ tn, float* __restrict__ x0) {
    const int idx = blockIdx.x * blockDim.x + threadIdx.x;
    if (idx >= NNODES * 256) return;
    const int c = idx & 255, node = idx >> 8;
    const int b = node / 18, i = node % 18;
    float v;
    if (i < 8)       v = vn[(b * 8 + i) * 256 + c];
    else if (i < 16) v = dn[(b * 8 + i - 8) * 256 + c];
    else if (i == 16) v = tn[b * 256 + c];
    else {
        float s = 0.f;
        for (int f2 = 0; f2 < 8; ++f2)
            s += vn[(b * 8 + f2) * 256 + c] + dn[(b * 8 + f2) * 256 + c];
        v = tn[b * 256 + c] + s * 0.125f;
    }
    x0[idx] = v;
}

// build hyperedge incidence (deterministic, tiny)
__global__ void k_init_edges(int* __restrict__ row, int* __restrict__ col) {
    if (threadIdx.x || blockIdx.x) return;
    int ec = 0, p = 0;
    for (int b = 0; b < 2; ++b) {
        const int s = b * 18;
        for (int i = 0; i < 8;  ++i) { row[p] = s + i;     col[p] = ec; ++p; } ++ec;  // v
        for (int i = 0; i < 8;  ++i) { row[p] = s + 8 + i; col[p] = ec; ++p; } ++ec;  // d
        for (int r2 = 0; r2 < 2; ++r2) {
            for (int i = 0; i < 16; ++i) { row[p] = s + i; col[p] = ec; ++p; } ++ec;  // v+d (x2)
        }
        for (int i = 0; i < 16; ++i) { row[p] = s + i; col[p] = ec; ++p; }
        row[p] = s + 16; col[p] = ec; ++p; ++ec;                                      // v+d+t
        row[p] = s + 16; col[p] = ec; ++p; ++ec;                                      // t
        row[p] = s + 17; col[p] = ec; ++p; ++ec;                                      // g
    }
}

// generic small dense: out[i][j] = act(sum_k in[i][k]*W[j][k])
__global__ void k_linear(const float* __restrict__ in, const float* __restrict__ W,
                         float* __restrict__ out, int n, int inC, int outF, int act) {
    const int idx = blockIdx.x * blockDim.x + threadIdx.x;
    if (idx >= n * outF) return;
    const int j = idx % outF, i = idx / outF;
    const float* a = in + (size_t)i * inC;
    const float* w = W + (size_t)j * inC;
    float s = 0.f;
    for (int k = 0; k < inC; ++k) s += a[k] * w[k];
    if (act == 1) s = s > 0.f ? s : 0.f;
    else if (act == 2) s = 1.f / (1.f + expf(-s));
    out[idx] = s;
}

// per-entry attention logits + leaky-relu(0.2)
__global__ void k_att(const float* __restrict__ xl, const float* __restrict__ ea,
                      const float* __restrict__ att, const int* __restrict__ row,
                      const int* __restrict__ col, float* __restrict__ aw, int outC) {
    const int idx = blockIdx.x * blockDim.x + threadIdx.x;
    if (idx >= NENT * 4) return;
    const int h = idx & 3, p = idx >> 2;
    const float* xr = xl + ((size_t)row[p] * 4 + h) * outC;
    const float* er = ea + ((size_t)col[p] * 4 + h) * outC;
    const float* at = att + (size_t)h * 2 * outC;
    float s = 0.f;
    for (int c = 0; c < outC; ++c) s += xr[c] * at[c] + er[c] * at[outC + c];
    aw[p * 4 + h] = s > 0.f ? s : 0.2f * s;
}

// softmax over edge segments + degree normalizers (tiny -> single thread)
__global__ void k_att_norm(const int* __restrict__ row, const int* __restrict__ col,
                           const float* __restrict__ edge_w, float* __restrict__ aw,
                           float* __restrict__ Dinv, float* __restrict__ Binv) {
    if (threadIdx.x || blockIdx.x) return;
    float smax[NEDGE][4], ssum[NEDGE][4], cnt[NEDGE], Dsum[NNODES];
    for (int e = 0; e < NEDGE; ++e) { cnt[e] = 0.f;
        for (int h = 0; h < 4; ++h) { smax[e][h] = -1e30f; ssum[e][h] = 0.f; } }
    for (int i = 0; i < NNODES; ++i) Dsum[i] = 0.f;
    for (int p = 0; p < NENT; ++p) {
        const int c = col[p];
        cnt[c] += 1.f; Dsum[row[p]] += edge_w[c];
        for (int h = 0; h < 4; ++h) smax[c][h] = fmaxf(smax[c][h], aw[p * 4 + h]);
    }
    for (int p = 0; p < NENT; ++p) {
        const int c = col[p];
        for (int h = 0; h < 4; ++h) {
            const float v = expf(aw[p * 4 + h] - smax[c][h]);
            aw[p * 4 + h] = v; ssum[c][h] += v;
        }
    }
    for (int p = 0; p < NENT; ++p) {
        const int c = col[p];
        for (int h = 0; h < 4; ++h) aw[p * 4 + h] /= (ssum[c][h] + 1e-16f);
    }
    for (int e = 0; e < NEDGE; ++e)  Binv[e] = cnt[e]  > 0.f ? 1.f / cnt[e]  : 0.f;
    for (int i = 0; i < NNODES; ++i) Dinv[i] = Dsum[i] > 0.f ? 1.f / Dsum[i] : 0.f;
}

// edge features: e[edge][h][c] = Binv[edge] * sum_{col==edge} a*xl[row]
__global__ void k_efeat(const float* __restrict__ xl, const float* __restrict__ aw,
                        const int* __restrict__ row, const int* __restrict__ col,
                        const float* __restrict__ Binv, float* __restrict__ ef, int outC) {
    const int idx = blockIdx.x * blockDim.x + threadIdx.x;
    if (idx >= NEDGE * 4 * outC) return;
    const int c = idx % outC, t = idx / outC, h = t & 3, e = t >> 2;
    float s = 0.f;
    for (int p = 0; p < NENT; ++p)
        if (col[p] == e) s += aw[p * 4 + h] * xl[((size_t)row[p] * 4 + h) * outC + c];
    ef[idx] = Binv[e] * s;
}

// node out: o[i][c] = 0.25 * Dinv[i] * sum_h sum_{row==i} a*e[col]
__global__ void k_onode(const float* __restrict__ ef, const float* __restrict__ aw,
                        const int* __restrict__ row, const int* __restrict__ col,
                        const float* __restrict__ Dinv, float* __restrict__ out, int outC) {
    const int idx = blockIdx.x * blockDim.x + threadIdx.x;
    if (idx >= NNODES * outC) return;
    const int c = idx % outC, i = idx / outC;
    float s = 0.f;
    for (int p = 0; p < NENT; ++p)
        if (row[p] == i)
            for (int h = 0; h < 4; ++h)
                s += aw[p * 4 + h] * ef[((size_t)col[p] * 4 + h) * outC + c];
    out[idx] = 0.25f * Dinv[i] * s;
}

// relu then group-norm (32 groups), in place; one block per node, one thread/group
__global__ void k_gn_relu(float* __restrict__ x, int outC) {
    const int node = blockIdx.x, g = threadIdx.x;
    const int gs = outC / 32;
    float* p = x + (size_t)node * outC + g * gs;
    float mu = 0.f;
    for (int i = 0; i < gs; ++i) { float v = p[i]; v = v > 0.f ? v : 0.f; p[i] = v; mu += v; }
    mu /= gs;
    float var = 0.f;
    for (int i = 0; i < gs; ++i) { const float d = p[i] - mu; var += d * d; }
    var /= gs;
    const float inv = rsqrtf(var + 1e-5f);
    for (int i = 0; i < gs; ++i) p[i] = (p[i] - mu) * inv;
}

__global__ void k_colmean(const float* __restrict__ x, float* __restrict__ y,
                          int n, int outC) {
    const int c = blockIdx.x * blockDim.x + threadIdx.x;
    if (c >= outC) return;
    float s = 0.f;
    for (int i = 0; i < n; ++i) s += x[(size_t)i * outC + c];
    y[c] = s / n;
}

__global__ void k_se_scale(float* __restrict__ x, const float* __restrict__ t2,
                           int n, int outC) {
    const int idx = blockIdx.x * blockDim.x + threadIdx.x;
    if (idx < n * outC) x[idx] *= t2[idx % outC];
}

// observers: max over windows of 4 of (tn + final observer node)
__global__ void k_observers(const float* __restrict__ tn, const float* __restrict__ xB,
                            float* __restrict__ out) {
    const int idx = blockIdx.x * blockDim.x + threadIdx.x;
    if (idx >= 128) return;
    const int b = idx >> 6, ch = idx & 63;
    float m = -1e30f;
    for (int q = 0; q < 4; ++q) {
        const int c = ch * 4 + q;
        m = fmaxf(m, tn[b * 256 + c] + xB[((size_t)(b * 18 + 17)) * 256 + c]);
    }
    out[idx] = m;
}

// decoder entry (all NHWC): vr2h = f16(v2 + vg broadcast over pixels)
__global__ void k_vr2(const float* __restrict__ v2, const float* __restrict__ xB,
                      _Float16* __restrict__ vr2h) {
    const size_t idx = (size_t)blockIdx.x * blockDim.x + threadIdx.x;
    if (idx >= (size_t)N_IMG * 1024 * 256) return;
    const int ch = (int)(idx & 255);
    const int img = (int)(idx >> 18);
    const int b = img >> 3, f2 = img & 7;
    vr2h[idx] = (_Float16)(v2[idx] + xB[((size_t)(b * 18 + f2)) * 256 + ch]);
}

// ---------------------------------------------------------------------------
static inline int ceil_div(long long a, int b) { return (int)((a + b - 1) / b); }

static void run_hg_layer(hipStream_t stream, const float* x_in, int inC, int outC,
                         const float* lin_w, const float* att, const float* edge_w,
                         const float* edge_attr, const float* se_w1, const float* se_w2,
                         const int* row, const int* col,
                         float* xl, float* ea, float* aw, float* Dinv, float* Binv,
                         float* ef, float* x_out, float* y0, float* t1, float* t2) {
    k_linear<<<ceil_div((long long)NNODES * 4 * outC, 256), 256, 0, stream>>>(
        x_in, lin_w, xl, NNODES, inC, 4 * outC, 0);
    k_linear<<<ceil_div((long long)NEDGE * 4 * outC, 256), 256, 0, stream>>>(
        edge_attr, lin_w, ea, NEDGE, inC, 4 * outC, 0);
    k_att<<<ceil_div(NENT * 4, 256), 256, 0, stream>>>(xl, ea, att, row, col, aw, outC);
    k_att_norm<<<1, 1, 0, stream>>>(row, col, edge_w, aw, Dinv, Binv);
    k_efeat<<<ceil_div((long long)NEDGE * 4 * outC, 256), 256, 0, stream>>>(
        xl, aw, row, col, Binv, ef, outC);
    k_onode<<<ceil_div((long long)NNODES * outC, 256), 256, 0, stream>>>(
        ef, aw, row, col, Dinv, x_out, outC);
    k_gn_relu<<<NNODES, 32, 0, stream>>>(x_out, outC);
    k_colmean<<<ceil_div(outC, 256), 256, 0, stream>>>(x_out, y0, NNODES, outC);
    k_linear<<<1, 256, 0, stream>>>(y0, se_w1, t1, 1, outC, outC / 16, 1);
    k_linear<<<ceil_div(outC, 256), 256, 0, stream>>>(t1, se_w2, t2, 1, outC / 16, outC, 2);
    k_se_scale<<<ceil_div((long long)NNODES * outC, 256), 256, 0, stream>>>(
        x_out, t2, NNODES, outC);
}

extern "C" void kernel_launch(void* const* d_in, const int* in_sizes, int n_in,
                              void* d_out_v, int out_size, void* d_ws, size_t ws_size,
                              hipStream_t stream) {
    (void)in_sizes; (void)n_in; (void)out_size; (void)ws_size;
    const float* vf = (const float*)d_in[0];  const float* vp = (const float*)d_in[1];
    const float* mf = (const float*)d_in[2];  const float* mp = (const float*)d_in[3];
    const float* tf = (const float*)d_in[4];  const float* tp = (const float*)d_in[5];
    const float* cv1_w = (const float*)d_in[6];  const float* cv1_b = (const float*)d_in[7];
    const float* cv2_w = (const float*)d_in[8];  const float* cv2_b = (const float*)d_in[9];
    const float* dc1_w = (const float*)d_in[10]; const float* dc1_b = (const float*)d_in[11];
    const float* dc2_w = (const float*)d_in[12]; const float* dc2_b = (const float*)d_in[13];
    const float* tc1_w = (const float*)d_in[14]; const float* tc1_b = (const float*)d_in[15];
    const float* tc2_w = (const float*)d_in[16]; const float* tc2_b = (const float*)d_in[17];
    const float* lin1_w = (const float*)d_in[18]; const float* att1 = (const float*)d_in[19];
    const float* lin2_w = (const float*)d_in[20]; const float* att2 = (const float*)d_in[21];
    const float* ew1 = (const float*)d_in[22]; const float* ew2 = (const float*)d_in[23];
    const float* ea1 = (const float*)d_in[24]; const float* ea2 = (const float*)d_in[25];
    const float* se1_w1 = (const float*)d_in[26]; const float* se1_w2 = (const float*)d_in[27];
    const float* se2_w1 = (const float*)d_in[28]; const float* se2_w2 = (const float*)d_in[29];
    float* d_out = (float*)d_out_v;

    char* ws = (char*)d_ws; size_t off = 0;
    auto A = [&](size_t bytes) -> char* {
        char* p = ws + off; off += bytes; off = (off + 255) & ~(size_t)255; return p;
    };
    const size_t S0 = (size_t)N_IMG * C0 * 128 * 128;   // 16.78M elems
    _Float16* v0h = (_Float16*)A(S0 * 2);               // 33.5 MB, NHWC (reused: r1h... no, see below)
    _Float16* m0h = (_Float16*)A(S0 * 2);               // 33.5 MB, NHWC (reused: m2 f32)
    float*    v1  = (float*)   A(S0 / 2 * 4);           // 33.5 MB, NHWC
    _Float16* v1h = (_Float16*)A(S0 / 2 * 2);           // 16.8 MB, NHWC (reused: r1h)
    float*    v2  = (float*)   A(S0 / 4 * 4);           // 16.8 MB, NHWC
    _Float16* m1h = (_Float16*)A(S0 / 2 * 2);           // 16.8 MB, NHWC (reused: vr2h)
    float*    m2   = (float*)m0h;                       // NHWC
    _Float16* r1h  = v1h;                               // NHWC
    _Float16* vr2h = m1h;                               // NHWC
    _Float16* ap_cv1 = (_Float16*)A(8  * 9 * 2 * 512 * 2);
    _Float16* ap_cv2 = (_Float16*)A(16 * 9 * 4 * 512 * 2);
    _Float16* ap_dc1 = (_Float16*)A(8  * 9 * 2 * 512 * 2);
    _Float16* ap_dc2 = (_Float16*)A(16 * 9 * 4 * 512 * 2);
    _Float16* ap_tc1 = (_Float16*)A(8  * 9 * 8 * 512 * 2);
    _Float16* ap_tc2 = (_Float16*)A(4  * 9 * 4 * 512 * 2);
    float* vn = (float*)A(16 * 256 * 4);
    float* dn = (float*)A(16 * 256 * 4);
    float* tn = (float*)A(2 * 256 * 4);
    float* x0 = (float*)A(NNODES * 256 * 4);
    float* xA = (float*)A(NNODES * 512 * 4);
    float* xB = (float*)A(NNODES * 256 * 4);
    float* xl = (float*)A(NNODES * 2048 * 4);
    float* eaL = (float*)A(NEDGE * 2048 * 4);
    float* ef  = (float*)A(NEDGE * 2048 * 4);
    float* aw  = (float*)A(NENT * 4 * 4);
    float* Dinv = (float*)A(NNODES * 4);
    float* Binv = (float*)A(NEDGE * 4);
    float* y0 = (float*)A(512 * 4);
    float* t1 = (float*)A(32 * 4);
    float* t2 = (float*)A(512 * 4);
    int* e_row = (int*)A(NENT * 4);
    int* e_col = (int*)A(NENT * 4);

    // 1. f32 NCHW inputs -> f16 NHWC staging (LDS transpose)
    k_addcast_nhwc<<<dim3(256, 16), 256, 0, stream>>>(vf, vp, v0h, 128 * 128);
    k_addcast_nhwc<<<dim3(256, 16), 256, 0, stream>>>(mf, mp, m0h, 128 * 128);

    // 2. weight prepacks into A-fragment layout
    k_prepack<0><<<144, 32, 0, stream>>>(cv1_w, ap_cv1, 64, 128);
    k_prepack<0><<<576, 32, 0, stream>>>(cv2_w, ap_cv2, 128, 256);
    k_prepack<0><<<144, 32, 0, stream>>>(dc1_w, ap_dc1, 64, 128);
    k_prepack<0><<<576, 32, 0, stream>>>(dc2_w, ap_dc2, 128, 256);
    k_prepack<1><<<576, 32, 0, stream>>>(tc1_w, ap_tc1, 256, 128);
    k_prepack<1><<<144, 32, 0, stream>>>(tc2_w, ap_tc2, 128, 64);

    // 3-6. encoder convs (NHWC; vision keeps f32 for residual/pool)
    k_conv_wmma<0, 0><<<dim3(64, 4, 16), 32, 0, stream>>>(
        v0h, ap_cv1, cv1_b, nullptr, nullptr, v1, v1h, 64, 128, 128, 128, 64, 64);
    k_conv_wmma<0, 0><<<dim3(16, 8, 16), 32, 0, stream>>>(
        v1h, ap_cv2, cv2_b, nullptr, nullptr, v2, nullptr, 128, 256, 64, 64, 32, 32);
    k_conv_wmma<0, 0><<<dim3(64, 4, 16), 32, 0, stream>>>(
        m0h, ap_dc1, dc1_b, nullptr, nullptr, nullptr, m1h, 64, 128, 128, 128, 64, 64);
    k_conv_wmma<0, 0><<<dim3(16, 8, 16), 32, 0, stream>>>(
        m1h, ap_dc2, dc2_b, nullptr, nullptr, m2, nullptr, 128, 256, 64, 64, 32, 32);

    // 7-9. pooling, text, node assembly
    k_pool_ln<<<16, 256, 0, stream>>>(v2, vn, 1024);
    k_pool_ln<<<16, 256, 0, stream>>>(m2, dn, 1024);
    k_tn<<<2, 256, 0, stream>>>(tf, tp, tn);
    k_init_edges<<<1, 1, 0, stream>>>(e_row, e_col);
    k_build_x<<<36, 256, 0, stream>>>(vn, dn, tn, x0);

    // 10-11. hypergraph layers (tiny)
    run_hg_layer(stream, x0, 256, 512, lin1_w, att1, ew1, ea1, se1_w1, se1_w2,
                 e_row, e_col, xl, eaL, aw, Dinv, Binv, ef, xA, y0, t1, t2);
    run_hg_layer(stream, xA, 512, 256, lin2_w, att2, ew2, ea2, se2_w1, se2_w2,
                 e_row, e_col, xl, eaL, aw, Dinv, Binv, ef, xB, y0, t1, t2);

    // 12. observers -> tail of d_out
    k_observers<<<1, 128, 0, stream>>>(tn, xB, d_out + S0);

    // 13-15. decoder: vr2 = v2+vg; r1 = v1 + relu(convT1); out = (vf+vp) + relu(convT2)
    k_vr2<<<(unsigned)((S0 / 4 + 255) / 256), 256, 0, stream>>>(v2, xB, vr2h);
    k_conv_wmma<1, 0><<<dim3(64, 4, 16), 32, 0, stream>>>(
        vr2h, ap_tc1, tc1_b, v1, nullptr, nullptr, r1h, 256, 128, 32, 32, 64, 64);
    k_conv_wmma<1, 1><<<dim3(256, 2, 16), 32, 0, stream>>>(
        r1h, ap_tc2, tc2_b, vf, vp, d_out, nullptr, 128, 64, 64, 64, 128, 128);
}
